// MoELayer_51986284151414
// MI455X (gfx1250) — matrix operand order repaired and checked by reading, not compiled, not run on hardware
//
#include <hip/hip_runtime.h>

// ---------------- problem constants ----------------
static constexpr int T = 16384;   // B*S tokens
static constexpr int D = 768;
static constexpr int F = 3072;
static constexpr int E = 8;

// ---------------- types ----------------
typedef __attribute__((ext_vector_type(16))) __bf16       v16bf;
typedef __attribute__((ext_vector_type(8)))  float        v8f;
typedef __attribute__((ext_vector_type(4)))  unsigned int u32x4;
typedef __attribute__((ext_vector_type(4)))  float        f32x4;
typedef __attribute__((ext_vector_type(2)))  unsigned int u32x2;

union FragU { u32x4 q[2]; v16bf v; };

__device__ __forceinline__ unsigned short f2bf(float f) {
    unsigned int u = __builtin_bit_cast(unsigned int, f);
    unsigned int r = u + 0x7fffu + ((u >> 16) & 1u);   // round-to-nearest-even
    return (unsigned short)(r >> 16);
}
__device__ __forceinline__ unsigned int f2bf2(float a, float b) {
    return (unsigned int)f2bf(a) | ((unsigned int)f2bf(b) << 16);
}

__device__ __forceinline__ v8f wmma_bf16(v16bf a, v16bf b, v8f c) {
    return __builtin_amdgcn_wmma_f32_16x16x32_bf16(false, a, false, b, (short)0, c, false, false);
}

// A fragment (16x32 bf16) from row-major LDS [16][ld]:
// lane: row = lane&15; K = kb..kb+15, kb = 16*(lane>>4).
__device__ __forceinline__ v16bf loadA(const unsigned short* base, int ld, int k0, int lane) {
    int row = lane & 15;
    int kh  = (lane >> 4) * 16;
    const unsigned short* p = base + row * ld + k0 + kh;
    FragU f;
    f.q[0] = *(const u32x4*)(p);
    f.q[1] = *(const u32x4*)(p + 8);
    return f.v;
}

// B fragment (32x16 bf16) direct from global, K-contiguous-per-column layout:
// caller points p at (col = base + lane&15)*ldK + 16*(lane>>4); 16 contiguous K.
__device__ __forceinline__ v16bf loadB_g(const unsigned short* p) {
    FragU f;
    f.q[0] = *(const u32x4*)(p);
    f.q[1] = *(const u32x4*)(p + 8);
    return f.v;
}

// ---------------- weight convert+transpose: f32 [R][C] -> bf16 [C][R] ----------------
__global__ __launch_bounds__(256) void moe_wconv_kernel(
    const float* __restrict__ w_in, const float* __restrict__ w_out,
    unsigned short* __restrict__ wtin, unsigned short* __restrict__ wtout)
{
    const int e  = blockIdx.y;
    const int zz = blockIdx.z;                 // 0: w_in (D x F), 1: w_out (F x D)
    const int Rr = zz == 0 ? D : F;
    const int Cc = zz == 0 ? F : D;
    const float* src = (zz == 0 ? w_in : w_out) + (size_t)e * D * F;
    unsigned short* dst = (zz == 0 ? wtin : wtout) + (size_t)e * D * F;

    const int tilesC = Cc / 64;
    const int ty = blockIdx.x / tilesC, tx = blockIdx.x % tilesC;
    const int r0 = ty * 64, c0 = tx * 64;

    __shared__ unsigned short tl[64 * 72];
    for (int idx = threadIdx.x; idx < 4096; idx += 256) {
        int ri = idx >> 6, ci = idx & 63;
        tl[ri * 72 + ci] = f2bf(src[(size_t)(r0 + ri) * Cc + c0 + ci]);
    }
    __syncthreads();
    for (int idx = threadIdx.x; idx < 4096; idx += 256) {
        int co = idx >> 6, ro = idx & 63;
        dst[(size_t)(c0 + co) * Rr + r0 + ro] = tl[ro * 72 + co];
    }
}

// ---------------- router kernel ----------------
__global__ __launch_bounds__(256) void moe_router_kernel(
    const float* __restrict__ x, const float* __restrict__ rw,
    float* __restrict__ logits_out, int* __restrict__ eidx_out,
    float* __restrict__ prob_out, int* __restrict__ cnt, int* __restrict__ bucket)
{
    __shared__ float srw[D * E];
    for (int i = threadIdx.x; i < D * E; i += 256) srw[i] = rw[i];
    __syncthreads();

    const int lane = threadIdx.x & 31;
    const int wave = blockIdx.x * 8 + (threadIdx.x >> 5);  // 64 blocks * 8 waves
    const int TPW  = T / 512;

    for (int it = 0; it < TPW; ++it) {
        int t = wave * TPW + it;
        float acc[E];
#pragma unroll
        for (int e = 0; e < E; ++e) acc[e] = 0.0f;
        for (int d = lane; d < D; d += 32) {
            float xv = x[(size_t)t * D + d];
#pragma unroll
            for (int e = 0; e < E; ++e) acc[e] += xv * srw[d * E + e];
        }
#pragma unroll
        for (int e = 0; e < E; ++e) {
#pragma unroll
            for (int off = 16; off >= 1; off >>= 1)
                acc[e] += __shfl_xor(acc[e], off, 32);
        }
        if (lane == 0) {
            float m = acc[0]; int mi = 0;
#pragma unroll
            for (int e = 1; e < E; ++e) if (acc[e] > m) { m = acc[e]; mi = e; }
            float s = 0.0f;
#pragma unroll
            for (int e = 0; e < E; ++e) s += __expf(acc[e] - m);
            float p = 1.0f / s;
#pragma unroll
            for (int e = 0; e < E; ++e) logits_out[(size_t)t * E + e] = acc[e];
            eidx_out[t] = mi;
            prob_out[t] = p;
            int slot = atomicAdd(&cnt[mi], 1);
            bucket[mi * T + slot] = t;
        }
    }
}

// ---------------- expert FFN kernel ----------------
static constexpr int TM  = 32;        // tokens per block (2 M-tiles)
static constexpr int LDX = D + 8;     // 776
static constexpr int LDH = 128 + 8;   // 136

__global__ __launch_bounds__(256) void moe_ffn_kernel(
    const float* __restrict__ x,
    const unsigned short* __restrict__ wtin,    // bf16 [E][F][D] (K=D contiguous)
    const unsigned short* __restrict__ wtout,   // bf16 [E][D][F] (K=F contiguous)
    const float* __restrict__ prob,
    const int* __restrict__ cnt, const int* __restrict__ bucket,
    float* __restrict__ out)
{
    const int e    = blockIdx.y;
    const int tile = blockIdx.x;
    const int c    = cnt[e];
    if (tile * TM >= c) return;

    const int lane = threadIdx.x & 31;
    const int w    = threadIdx.x >> 5;
    const int cc   = lane & 15;
    const int lh   = lane >> 4;

    __shared__ unsigned short sX[TM * LDX];   // 49.7 KB bf16 x tile
    __shared__ unsigned short sH[TM * LDH];   //  8.7 KB relu(h) chunk
    __shared__ int   sTok[TM];
    __shared__ float sProb[TM];
    __shared__ int   sValid[TM];

    if (threadIdx.x < TM) {
        int s  = tile * TM + (int)threadIdx.x;
        int sc = s < c ? s : c - 1;
        int t  = bucket[e * T + sc];
        sTok[threadIdx.x]   = t;
        sProb[threadIdx.x]  = (s < c) ? prob[t] : 0.0f;
        sValid[threadIdx.x] = (s < c) ? 1 : 0;
    }
    __syncthreads();

    // stage x tile (gather rows, f32 -> bf16, 4-wide)
    for (int q = threadIdx.x; q < TM * (D / 4); q += 256) {
        int r  = q / (D / 4);
        int cq = (q % (D / 4)) * 4;
        f32x4 xv = *(const f32x4*)(x + (size_t)sTok[r] * D + cq);
        u32x2 pk;
        pk.x = f2bf2(xv.x, xv.y);
        pk.y = f2bf2(xv.z, xv.w);
        *(u32x2*)(sX + r * LDX + cq) = pk;
    }
    __syncthreads();

    const unsigned short* WinT  = wtin  + (size_t)e * F * D;
    const unsigned short* WoutT = wtout + (size_t)e * D * F;

    v8f acc[2][6];
#pragma unroll
    for (int m = 0; m < 2; ++m)
#pragma unroll
        for (int j = 0; j < 6; ++j)
            acc[m][j] = (v8f){0.f,0.f,0.f,0.f,0.f,0.f,0.f,0.f};

    for (int fb = 0; fb < F; fb += 128) {
        // ---- step 1: h[32,128] = relu(x @ Win[:, fb:fb+128]); wave w owns cols [16w,16w+16)
        v8f h0 = (v8f){0.f,0.f,0.f,0.f,0.f,0.f,0.f,0.f};
        v8f h1 = (v8f){0.f,0.f,0.f,0.f,0.f,0.f,0.f,0.f};
        const unsigned short* b1p = WinT + (size_t)(fb + w * 16 + cc) * D + lh * 16;
#pragma unroll 8
        for (int k0 = 0; k0 < D; k0 += 32) {
            v16bf b  = loadB_g(b1p + k0);
            v16bf a0 = loadA(sX,            LDX, k0, lane);
            v16bf a1 = loadA(sX + 16 * LDX, LDX, k0, lane);
            h0 = wmma_bf16(a0, b, h0);
            h1 = wmma_bf16(a1, b, h1);
        }
        __syncthreads();   // previous chunk's sH readers are done
        {
            int coln = w * 16 + cc;
#pragma unroll
            for (int v = 0; v < 8; ++v) {
                float z0 = h0[v] > 0.f ? h0[v] : 0.f;
                float z1 = h1[v] > 0.f ? h1[v] : 0.f;
                sH[(v + 8 * lh) * LDH + coln]        = f2bf(z0);
                sH[(16 + v + 8 * lh) * LDH + coln]   = f2bf(z1);
            }
        }
        __syncthreads();

        // hoist all h A-fragments for this chunk (reused across all 6 N sub-tiles)
        v16bf aF[4][2];
#pragma unroll
        for (int ks = 0; ks < 4; ++ks) {
            aF[ks][0] = loadA(sH,            LDH, ks * 32, lane);
            aF[ks][1] = loadA(sH + 16 * LDH, LDH, ks * 32, lane);
        }

        // ---- step 2: y[32,768] += h @ Wout[fb:fb+128, :]; wave w owns col 16w+cc of each 128-sub
#pragma unroll
        for (int j = 0; j < 6; ++j) {
            const unsigned short* b2p =
                WoutT + (size_t)(j * 128 + w * 16 + cc) * F + fb + lh * 16;
#pragma unroll
            for (int ks = 0; ks < 4; ++ks) {
                v16bf b = loadB_g(b2p + ks * 32);
                acc[0][j] = wmma_bf16(aF[ks][0], b, acc[0][j]);
                acc[1][j] = wmma_bf16(aF[ks][1], b, acc[1][j]);
            }
        }
    }

    // ---- scaled scatter store: out[token, col] = top_prob * y
#pragma unroll
    for (int m = 0; m < 2; ++m) {
#pragma unroll
        for (int v = 0; v < 8; ++v) {
            int r = m * 16 + v + 8 * lh;
            if (sValid[r]) {
                float  ps = sProb[r];
                size_t tb = (size_t)sTok[r] * D;
#pragma unroll
                for (int j = 0; j < 6; ++j)
                    out[tb + j * 128 + w * 16 + cc] = ps * acc[m][j][v];
            }
        }
    }
}

// ---------------- launcher ----------------
extern "C" void kernel_launch(void* const* d_in, const int* in_sizes, int n_in,
                              void* d_out, int out_size, void* d_ws, size_t ws_size,
                              hipStream_t stream) {
    (void)in_sizes; (void)n_in; (void)out_size; (void)ws_size;
    const float* x    = (const float*)d_in[0];  // [B,S,D]
    const float* rw   = (const float*)d_in[1];  // [D,E]
    const float* win  = (const float*)d_in[2];  // [E,D,F]
    const float* wout = (const float*)d_in[3];  // [E,F,D]

    float* out    = (float*)d_out;                       // [B,S,D]
    float* logits = out + (size_t)T * D;                 // [T,E]
    int*   eidx   = (int*)(logits + (size_t)T * E);      // [T]

    int*   cnt    = (int*)d_ws;                          // [E]
    int*   bucket = cnt + E;                             // [E,T]
    float* prob   = (float*)(bucket + (size_t)E * T);    // [T]
    unsigned short* wtin  = (unsigned short*)(prob + T); // bf16 [E][F][D]
    unsigned short* wtout = wtin + (size_t)E * D * F;    // bf16 [E][D][F]

    hipMemsetAsync(cnt, 0, E * sizeof(int), stream);
    moe_wconv_kernel<<<dim3(576, E, 2), 256, 0, stream>>>(win, wout, wtin, wtout);
    moe_router_kernel<<<64, 256, 0, stream>>>(x, rw, logits, eidx, prob, cnt, bucket);
    moe_ffn_kernel<<<dim3(T / TM, E), 256, 0, stream>>>(x, wtin, wtout, prob, cnt, bucket, out);
}
